// up_fcnHead_51402168598690
// MI455X (gfx1250) — matrix operand order repaired and checked by reading, not compiled, not run on hardware
//
#include <hip/hip_runtime.h>

typedef __attribute__((ext_vector_type(16))) _Float16 v16h;
typedef __attribute__((ext_vector_type(8)))  _Float16 v8h;
typedef __attribute__((ext_vector_type(8)))  float    v8f;

#define BN_EPS 1e-5f

// ---------- elementwise f32 -> f16 ----------
__global__ void k_cvt(const float* __restrict__ s, _Float16* __restrict__ d, int n) {
  int i = blockIdx.x * 256 + threadIdx.x;
  if (i < n) d[i] = (_Float16)s[i];
}

// ---------- weights (Co,Ci,taps) -> fragment-ready f16 layout ----------
// For each (tap, kc, cotile): 1KB block = [lane 0..31][e 0..15] halfs, where
// lane=(half,row): element e holds B[K = half*16+e][co = cotile*16+row],
// ci = kc*32+K. This is exactly the per-lane v16h B fragment for WMMA, so the
// GEMM can async-copy blocks to LDS with no transpose and read fragments as
// one contiguous 32B run per lane. co >= Co zero-padded.
__global__ void k_wt(const float* __restrict__ w, _Float16* __restrict__ wt,
                     int Co, int Ci, int taps, int CoPad) {
  int i = blockIdx.x * 256 + threadIdx.x;
  int tot = taps * Ci * CoPad;
  if (i >= tot) return;
  int e    = i & 15;
  int lane = (i >> 4) & 31;
  int blk  = i >> 9;
  int nct  = CoPad >> 4;
  int cotile = blk % nct;
  int tq     = blk / nct;
  int kcN  = Ci >> 5;
  int kc   = tq % kcN;
  int tap  = tq / kcN;
  int half = lane >> 4, rowl = lane & 15;
  int K    = half * 16 + e;
  int ci   = (kc << 5) + K;
  int co   = (cotile << 4) + rowl;
  float v = (co < Co) ? w[(co * Ci + ci) * taps + tap] : 0.f;
  wt[i] = (_Float16)v;
}

// ---------- folded BN scale: s = g * rsqrt(1+eps); g==null -> 1.0 ----------
__global__ void k_scale(const float* __restrict__ g, float* __restrict__ s, int n) {
  int i = blockIdx.x * 256 + threadIdx.x;
  if (i < n) s[i] = g ? g[i] * rsqrtf(1.f + BN_EPS) : 1.f;
}

// ---------- bilinear upsample, align_corners=True, f16 -> f16 ----------
__global__ void k_up(const _Float16* __restrict__ in, _Float16* __restrict__ out,
                     int NC, int IH, int IW, int OH, int OW) {
  int i = blockIdx.x * 256 + threadIdx.x;
  int tot = NC * OH * OW;
  if (i >= tot) return;
  int nc = i / (OH * OW);
  int r  = i % (OH * OW);
  int oy = r / OW, ox = r % OW;
  float sy = (float)(IH - 1) / (float)(OH - 1);
  float sx = (float)(IW - 1) / (float)(OW - 1);
  float fy = oy * sy, fx = ox * sx;
  int y0 = (int)fy, x0 = (int)fx;
  int y1 = min(y0 + 1, IH - 1), x1 = min(x0 + 1, IW - 1);
  float ty = fy - (float)y0, tx = fx - (float)x0;
  const _Float16* p = in + (long)nc * IH * IW;
  float v00 = (float)p[y0 * IW + x0], v01 = (float)p[y0 * IW + x1];
  float v10 = (float)p[y1 * IW + x0], v11 = (float)p[y1 * IW + x1];
  float v = v00 * (1.f - ty) * (1.f - tx) + v01 * (1.f - ty) * tx +
            v10 * ty * (1.f - tx) + v11 * ty * tx;
  out[i] = (_Float16)v;
}

// ---------- generic WMMA conv GEMM ----------
// im2col GEMM: M = NB*H*W positions, N = CoPad channels, K = taps*Ci.
// Block = 256 threads (8 waves). Tile = 128(M) x 64(N). Double-buffered LDS,
// software pipelined (one barrier per K-chunk). A (im2col w/ zero halo) is
// staged through VGPRs; B (weights, fragment-ready global layout) is staged
// with the CDNA5 async engine: one global_load_async_to_lds_b128 per thread
// per chunk, gated by s_wait_asynccnt.
__global__ void __launch_bounds__(256)
k_conv_wmma(const _Float16* __restrict__ in, const _Float16* __restrict__ wt,
            const float* __restrict__ scale, const float* __restrict__ bias,
            _Float16* __restrict__ outH, float* __restrict__ outF,
            int Ci, int H, int W, int CoPad, int CoStore,
            int taps, int dil, int relu) {
  __shared__ _Float16 As[2][128][40];   // [buf][m][k], 80B row stride
  __shared__ _Float16 BsF[2][2048];     // [buf][4 cotiles x 32 lanes x 16 halfs]

  const int t    = threadIdx.x;
  const int wv   = t >> 5;
  const int lane = t & 31;
  const int half = lane >> 4;
  const int row  = lane & 15;
  const int HW   = H * W;
  const int m0   = blockIdx.x * 128;
  const int n0   = blockIdx.y * 64;

  // A staging role (invariant): thread stages row mst, k phase tA
  const int mst = t & 127;
  const int tA  = t >> 7;     // 0/1
  const int pst = m0 + mst;
  const int nst = pst / HW;
  const int rst = pst % HW;
  const int yst = rst / W;
  const int xst = rst % W;

  const int kcN   = Ci >> 5;
  const int total = taps * kcN;
  const int nct   = CoPad >> 4;
  const int ct0   = n0 >> 4;

  auto stage = [&](int tap, int kc, int buf) {
    // --- B: async copy of 4KB fragment-ready block (16B per thread) ---
    {
      const _Float16* bsrc =
          wt + ((long)((tap * kcN + kc) * nct + ct0) * 512 + (long)t * 8);
      unsigned ldsoff = (unsigned)(size_t)&BsF[buf][t * 8];
      unsigned long long ga = (unsigned long long)(size_t)bsrc;
      asm volatile("global_load_async_to_lds_b128 %0, %1, off"
                   :: "v"(ldsoff), "v"(ga) : "memory");
    }
    // --- A: implicit im2col with zero halo, through VGPRs ---
    const int dy = (taps == 9) ? dil * ((tap / 3) - 1) : 0;
    const int dx = (taps == 9) ? dil * ((tap % 3) - 1) : 0;
    const int iy = yst + dy, ix = xst + dx;
    const bool inb = (iy >= 0 && iy < H && ix >= 0 && ix < W);
    const _Float16* arow = in + ((long)nst * Ci + (kc << 5)) * HW + iy * W + ix;
    if (inb && kc + 1 < kcN) __builtin_prefetch(arow + (long)32 * HW, 0, 1);
#pragma unroll
    for (int i = 0; i < 16; ++i) {
      int kk = i * 2 + tA;
      _Float16 v = (_Float16)0.f;
      if (inb) v = arow[(long)kk * HW];
      As[buf][mst][kk] = v;
    }
  };

  v8f acc0 = {}, acc1 = {}, acc2 = {}, acc3 = {};

  stage(0, 0, 0);
  int tap = 0, kc = 0;
  for (int u = 0; u < total; ++u) {
    // previous chunk's async batch must have landed before the barrier
    asm volatile("s_wait_asynccnt 0x0" ::: "memory");
    __syncthreads();
    int ntap = tap, nkc = kc + 1;
    if (nkc == kcN) { nkc = 0; ntap++; }
    if (u + 1 < total) stage(ntap, nkc, (u + 1) & 1);

    const int buf = u & 1;
    // A fragment: lane (half,row) needs K = half*16+0..15
    const _Float16* ar = &As[buf][wv * 16 + row][half * 16];
    v8h alo = *(const v8h*)ar;
    v8h ahi = *(const v8h*)(ar + 8);
    v16h a = __builtin_shufflevector(alo, ahi, 0, 1, 2, 3, 4, 5, 6, 7,
                                     8, 9, 10, 11, 12, 13, 14, 15);
    // B fragments: lane's own contiguous 32B run per cotile
    v16h b0, b1, b2, b3;
    {
      const _Float16* br = &BsF[buf][(0 * 32 + lane) * 16];
      v8h lo = *(const v8h*)br, hi = *(const v8h*)(br + 8);
      b0 = __builtin_shufflevector(lo, hi, 0, 1, 2, 3, 4, 5, 6, 7,
                                   8, 9, 10, 11, 12, 13, 14, 15);
    }
    {
      const _Float16* br = &BsF[buf][(1 * 32 + lane) * 16];
      v8h lo = *(const v8h*)br, hi = *(const v8h*)(br + 8);
      b1 = __builtin_shufflevector(lo, hi, 0, 1, 2, 3, 4, 5, 6, 7,
                                   8, 9, 10, 11, 12, 13, 14, 15);
    }
    {
      const _Float16* br = &BsF[buf][(2 * 32 + lane) * 16];
      v8h lo = *(const v8h*)br, hi = *(const v8h*)(br + 8);
      b2 = __builtin_shufflevector(lo, hi, 0, 1, 2, 3, 4, 5, 6, 7,
                                   8, 9, 10, 11, 12, 13, 14, 15);
    }
    {
      const _Float16* br = &BsF[buf][(3 * 32 + lane) * 16];
      v8h lo = *(const v8h*)br, hi = *(const v8h*)(br + 8);
      b3 = __builtin_shufflevector(lo, hi, 0, 1, 2, 3, 4, 5, 6, 7,
                                   8, 9, 10, 11, 12, 13, 14, 15);
    }

    acc0 = __builtin_amdgcn_wmma_f32_16x16x32_f16(false, a, false, b0, (short)0, acc0, false, false);
    acc1 = __builtin_amdgcn_wmma_f32_16x16x32_f16(false, a, false, b1, (short)0, acc1, false, false);
    acc2 = __builtin_amdgcn_wmma_f32_16x16x32_f16(false, a, false, b2, (short)0, acc2, false, false);
    acc3 = __builtin_amdgcn_wmma_f32_16x16x32_f16(false, a, false, b3, (short)0, acc3, false, false);

    tap = ntap; kc = nkc;
  }

  // writeback with fused BN (+ReLU); C/D layout: N=lane&15, M=(lane>>4)*8+r
#pragma unroll
  for (int nt = 0; nt < 4; ++nt) {
    v8f acc = (nt == 0) ? acc0 : (nt == 1) ? acc1 : (nt == 2) ? acc2 : acc3;
    int co = n0 + nt * 16 + row;
    if (co >= CoStore) continue;
    float sc = scale[co], bi = bias[co];
#pragma unroll
    for (int r8 = 0; r8 < 8; ++r8) {
      int p = m0 + wv * 16 + half * 8 + r8;
      int n = p / HW, rr = p % HW;
      float val = acc[r8] * sc + bi;
      if (relu) val = fmaxf(val, 0.f);
      long oidx = (long)(n * CoStore + co) * HW + rr;
      if (outH) outH[oidx] = (_Float16)val;
      else      outF[oidx] = val;
    }
  }
}

// ---------- energy (q . unfold(up_c2)) + softmax over 9 taps ----------
__global__ void k_att(const _Float16* __restrict__ q, const _Float16* __restrict__ key,
                      float* __restrict__ att, int NB) {
  int p = blockIdx.x * 256 + threadIdx.x;
  int tot = NB * 9216;
  if (p >= tot) return;
  int n = p / 9216, r = p % 9216, y = r / 96, x = r % 96;
  float e[9];
  int off[9];
  bool ok[9];
#pragma unroll
  for (int k = 0; k < 9; ++k) {
    int iy = y + 2 * ((k / 3) - 1), ix = x + 2 * ((k % 3) - 1);
    ok[k] = (iy >= 0 && iy < 96 && ix >= 0 && ix < 96);
    off[k] = iy * 96 + ix;
    e[k] = 0.f;
  }
  const _Float16* qb = q   + (long)n * 64 * 9216 + r;
  const _Float16* kb = key + (long)n * 64 * 9216;
  for (int c = 0; c < 64; ++c) {
    float qv = (float)qb[(long)c * 9216];
    const _Float16* kc = kb + (long)c * 9216;
#pragma unroll
    for (int k = 0; k < 9; ++k)
      if (ok[k]) e[k] += qv * (float)kc[off[k]];
  }
  float m = e[0];
#pragma unroll
  for (int k = 1; k < 9; ++k) m = fmaxf(m, e[k]);
  float s = 0.f;
#pragma unroll
  for (int k = 0; k < 9; ++k) { e[k] = __expf(e[k] - m); s += e[k]; }
  float inv = 1.f / s;
#pragma unroll
  for (int k = 0; k < 9; ++k) att[(long)p * 9 + k] = e[k] * inv;
}

// ---------- attention-weighted aggregation: out[c,p] = sum_k v96[c,nbr_k(p)]*att[p,k] ----------
__global__ void k_agg(const _Float16* __restrict__ v96, const float* __restrict__ att,
                      _Float16* __restrict__ out, int NB) {
  long i = (long)blockIdx.x * 256 + threadIdx.x;
  long tot = (long)NB * 512 * 9216;
  if (i >= tot) return;
  int n = (int)(i / (512L * 9216));
  int rem = (int)(i % (512L * 9216));
  int c = rem / 9216, r = rem % 9216, y = r / 96, x = r % 96;
  const float* ap = att + ((long)n * 9216 + r) * 9;
  const _Float16* vp = v96 + (long)(n * 512 + c) * 9216;
  float acc = 0.f;
#pragma unroll
  for (int k = 0; k < 9; ++k) {
    int iy = y + 2 * ((k / 3) - 1), ix = x + 2 * ((k % 3) - 1);
    if (iy >= 0 && iy < 96 && ix >= 0 && ix < 96)
      acc += ap[k] * (float)vp[iy * 96 + ix];
  }
  out[i] = (_Float16)acc;
}

extern "C" void kernel_launch(void* const* d_in, const int* in_sizes, int n_in,
                              void* d_out, int out_size, void* d_ws, size_t ws_size,
                              hipStream_t stream) {
  const float* c1   = (const float*)d_in[0];
  const float* c2   = (const float*)d_in[1];
  const float* x    = (const float*)d_in[2];
  const float* w_r1 = (const float*)d_in[3];
  const float* g_r1 = (const float*)d_in[4];
  const float* b_r1 = (const float*)d_in[5];
  const float* w_r2 = (const float*)d_in[6];
  const float* g_r2 = (const float*)d_in[7];
  const float* b_r2 = (const float*)d_in[8];
  const float* w_u  = (const float*)d_in[9];
  const float* g_u  = (const float*)d_in[10];
  const float* b_u  = (const float*)d_in[11];
  const float* w5   = (const float*)d_in[12];
  const float* g5   = (const float*)d_in[13];
  const float* b5   = (const float*)d_in[14];
  const float* w6   = (const float*)d_in[15];
  const float* b6   = (const float*)d_in[16];
  float* out = (float*)d_out;

  // workspace carve-up (256B aligned slabs)
  char* wsp = (char*)d_ws;
  auto alloc = [&](size_t bytes) -> void* {
    void* p = (void*)wsp;
    wsp += (bytes + 255) & ~(size_t)255;
    return p;
  };
  const int NB = 2;
  _Float16* c1h  = (_Float16*)alloc((size_t)NB * 256 * 9216 * 2);
  _Float16* xh   = (_Float16*)alloc((size_t)NB * 2048 * 2304 * 2);
  _Float16* c2h  = (_Float16*)alloc((size_t)NB * 512 * 2304 * 2);
  _Float16* c2u  = (_Float16*)alloc((size_t)NB * 512 * 9216 * 2);
  _Float16* q1   = (_Float16*)alloc((size_t)NB * 64 * 9216 * 2);
  _Float16* q2   = (_Float16*)alloc((size_t)NB * 64 * 9216 * 2);
  _Float16* upc2 = (_Float16*)alloc((size_t)NB * 64 * 9216 * 2);
  _Float16* v48  = (_Float16*)alloc((size_t)NB * 512 * 2304 * 2);
  _Float16* v96  = (_Float16*)alloc((size_t)NB * 512 * 9216 * 2);
  _Float16* agg  = (_Float16*)alloc((size_t)NB * 512 * 9216 * 2);
  _Float16* wt_r1 = (_Float16*)alloc((size_t)9 * 256 * 64 * 2);
  _Float16* wt_r2 = (_Float16*)alloc((size_t)9 * 64 * 64 * 2);
  _Float16* wt_u  = (_Float16*)alloc((size_t)512 * 64 * 2);
  _Float16* wt5   = (_Float16*)alloc((size_t)9 * 2048 * 512 * 2);
  _Float16* wt6   = (_Float16*)alloc((size_t)512 * 64 * 2);
  float* s_r1 = (float*)alloc(64 * 4);
  float* s_r2 = (float*)alloc(64 * 4);
  float* s_u  = (float*)alloc(64 * 4);
  float* s5   = (float*)alloc(512 * 4);
  float* s6   = (float*)alloc(64 * 4);
  float* att  = (float*)alloc((size_t)NB * 9216 * 9 * 4);

  auto gb = [](int n) { return dim3((n + 255) / 256); };

  // 1) convert activations
  k_cvt<<<gb(NB * 256 * 9216), 256, 0, stream>>>(c1, c1h, NB * 256 * 9216);
  k_cvt<<<gb(NB * 2048 * 2304), 256, 0, stream>>>(x, xh, NB * 2048 * 2304);
  k_cvt<<<gb(NB * 512 * 2304), 256, 0, stream>>>(c2, c2h, NB * 512 * 2304);

  // 2) convert weights to fragment-ready layout
  k_wt<<<gb(9 * 256 * 64), 256, 0, stream>>>(w_r1, wt_r1, 64, 256, 9, 64);
  k_wt<<<gb(9 * 64 * 64), 256, 0, stream>>>(w_r2, wt_r2, 64, 64, 9, 64);
  k_wt<<<gb(512 * 64), 256, 0, stream>>>(w_u, wt_u, 64, 512, 1, 64);
  k_wt<<<gb(9 * 2048 * 512), 256, 0, stream>>>(w5, wt5, 512, 2048, 9, 512);
  k_wt<<<gb(512 * 64), 256, 0, stream>>>(w6, wt6, 19, 512, 1, 64);

  // 3) folded BN scales
  k_scale<<<1, 256, 0, stream>>>(g_r1, s_r1, 64);
  k_scale<<<1, 256, 0, stream>>>(g_r2, s_r2, 64);
  k_scale<<<1, 256, 0, stream>>>(g_u, s_u, 64);
  k_scale<<<gb(512), 256, 0, stream>>>(g5, s5, 512);
  k_scale<<<1, 256, 0, stream>>>(nullptr, s6, 64);  // ones for conv6

  // 4) upsample c2 48->96
  k_up<<<gb(NB * 512 * 9216), 256, 0, stream>>>(c2h, c2u, NB * 512, 48, 48, 96, 96);

  // 5) refine convs (3x3 dil=2, BN+ReLU): 256->64, then 64->64 @96x96
  k_conv_wmma<<<dim3(144, 1), 256, 0, stream>>>(c1h, wt_r1, s_r1, b_r1, q1, nullptr,
                                                256, 96, 96, 64, 64, 9, 2, 1);
  k_conv_wmma<<<dim3(144, 1), 256, 0, stream>>>(q1, wt_r2, s_r2, b_r2, q2, nullptr,
                                                64, 96, 96, 64, 64, 9, 2, 1);
  // 6) refine2 1x1 512->64 on upsampled c2
  k_conv_wmma<<<dim3(144, 1), 256, 0, stream>>>(c2u, wt_u, s_u, b_u, upc2, nullptr,
                                                512, 96, 96, 64, 64, 1, 0, 1);
  // 7) conv5 3x3 2048->512 @48x48 (dominant GEMM)
  k_conv_wmma<<<dim3(36, 8), 256, 0, stream>>>(xh, wt5, s5, b5, v48, nullptr,
                                               2048, 48, 48, 512, 512, 9, 1, 1);
  // 8) upsample v 48->96
  k_up<<<gb(NB * 512 * 9216), 256, 0, stream>>>(v48, v96, NB * 512, 48, 48, 96, 96);

  // 9) attention: energy + softmax over 9 taps
  k_att<<<gb(NB * 9216), 256, 0, stream>>>(q2, upc2, att, NB);

  // 10) attention-weighted aggregation
  k_agg<<<gb(NB * 512 * 9216), 256, 0, stream>>>(v96, att, agg, NB);

  // 11) conv6 1x1 512->19 (+bias) -> d_out (fp32)
  k_conv_wmma<<<dim3(144, 1), 256, 0, stream>>>(agg, wt6, s6, b6, nullptr, out,
                                                512, 96, 96, 64, 19, 1, 0, 0);
}
